// MultiHeadAttention_26723286516321
// MI455X (gfx1250) — matrix operand order repaired and checked by reading, not compiled, not run on hardware
//
#include <hip/hip_runtime.h>
#include <hip/hip_bf16.h>

// ---------------- problem constants (match reference) ----------------
constexpr int CB = 128;   // batch
constexpr int CS = 256;   // seq
constexpr int CD = 512;   // head size (== model dim here)
constexpr int CH = 8;     // heads
constexpr int CHD = CH * CD;  // 4096

// ---------------- types ----------------
typedef __bf16 bf16_t;
typedef __attribute__((ext_vector_type(16))) __bf16 v16bf;
typedef __attribute__((ext_vector_type(8)))  __bf16 v8bf;
typedef __attribute__((ext_vector_type(8)))  float  v8f;

__device__ __forceinline__ bf16_t to_bf16(float f) {
  unsigned u = __builtin_bit_cast(unsigned, f);
  u += 0x7FFFu + ((u >> 16) & 1u);               // round to nearest even
  unsigned short h = (unsigned short)(u >> 16);
  return __builtin_bit_cast(bf16_t, h);
}

// A fragment (16x32 bf16, MxK): lane row = lane%16.
// elements 0..7  : K = k0 + e       + 8*hi
// elements 8..15 : K = k0 + 16+(e-8)+ 8*hi
__device__ __forceinline__ v16bf load_a_frag(const bf16_t* rowk, int hi) {
  v8bf lo = *(const v8bf*)(rowk + hi * 8);
  v8bf up = *(const v8bf*)(rowk + 16 + hi * 8);
  return __builtin_shufflevector(lo, up, 0,1,2,3,4,5,6,7,8,9,10,11,12,13,14,15);
}

// B fragment (32x16 bf16, KxN): lane col = lane%16, elements: same K order as A fragment.
// browk points at row-major-along-K storage of column n (i.e. W[n][k0...]).
__device__ __forceinline__ v16bf load_b_frag(const bf16_t* browk, int hi) {
  v8bf lo = *(const v8bf*)(browk + hi * 8);
  v8bf up = *(const v8bf*)(browk + 16 + hi * 8);
  return __builtin_shufflevector(lo, up, 0,1,2,3,4,5,6,7,8,9,10,11,12,13,14,15);
}

__device__ __forceinline__ v8f wmma_bf16(v16bf a, v16bf b, v8f c) {
  return __builtin_amdgcn_wmma_f32_16x16x32_bf16(false, a, false, b, (short)0, c,
                                                 false, false);
}

// ---- ping-pong pipelined 16(M)x64(N) wave GEMM (Kdim multiple of 64) ----
__device__ __forceinline__ void gemm16x64(const bf16_t* __restrict__ arow,
                                          const bf16_t* __restrict__ brow0,
                                          int ldb, int Kdim, int hi, v8f acc[4]) {
  v16bf a0 = load_a_frag(arow, hi);
  v16bf b0[4];
#pragma unroll
  for (int i = 0; i < 4; ++i) b0[i] = load_b_frag(brow0 + (size_t)i * 16 * ldb, hi);
  v16bf a1 = load_a_frag(arow + 32, hi);
  v16bf b1[4];
#pragma unroll
  for (int i = 0; i < 4; ++i)
    b1[i] = load_b_frag(brow0 + (size_t)i * 16 * ldb + 32, hi);
#pragma unroll 1
  for (int kk = 64; kk < Kdim; kk += 64) {
#pragma unroll
    for (int i = 0; i < 4; ++i) acc[i] = wmma_bf16(a0, b0[i], acc[i]);
    a0 = load_a_frag(arow + kk, hi);
#pragma unroll
    for (int i = 0; i < 4; ++i)
      b0[i] = load_b_frag(brow0 + (size_t)i * 16 * ldb + kk, hi);
#pragma unroll
    for (int i = 0; i < 4; ++i) acc[i] = wmma_bf16(a1, b1[i], acc[i]);
    a1 = load_a_frag(arow + kk + 32, hi);
#pragma unroll
    for (int i = 0; i < 4; ++i)
      b1[i] = load_b_frag(brow0 + (size_t)i * 16 * ldb + kk + 32, hi);
  }
#pragma unroll
  for (int i = 0; i < 4; ++i) acc[i] = wmma_bf16(a0, b0[i], acc[i]);
#pragma unroll
  for (int i = 0; i < 4; ++i) acc[i] = wmma_bf16(a1, b1[i], acc[i]);
}

// ---- ping-pong pipelined 32(M)x64(N) wave GEMM: 8 accums, 2 A-fragments ----
__device__ __forceinline__ void gemm32x64(const bf16_t* __restrict__ arow0,
                                          const bf16_t* __restrict__ arow1,
                                          const bf16_t* __restrict__ brow0,
                                          int ldb, int Kdim, int hi, v8f acc[8]) {
  v16bf a0lo = load_a_frag(arow0, hi);
  v16bf a0hi = load_a_frag(arow1, hi);
  v16bf b0[4];
#pragma unroll
  for (int i = 0; i < 4; ++i) b0[i] = load_b_frag(brow0 + (size_t)i * 16 * ldb, hi);
  v16bf a1lo = load_a_frag(arow0 + 32, hi);
  v16bf a1hi = load_a_frag(arow1 + 32, hi);
  v16bf b1[4];
#pragma unroll
  for (int i = 0; i < 4; ++i)
    b1[i] = load_b_frag(brow0 + (size_t)i * 16 * ldb + 32, hi);
#pragma unroll 1
  for (int kk = 64; kk < Kdim; kk += 64) {
#pragma unroll
    for (int i = 0; i < 4; ++i) {
      acc[i]     = wmma_bf16(a0lo, b0[i], acc[i]);
      acc[4 + i] = wmma_bf16(a0hi, b0[i], acc[4 + i]);
    }
    a0lo = load_a_frag(arow0 + kk, hi);
    a0hi = load_a_frag(arow1 + kk, hi);
#pragma unroll
    for (int i = 0; i < 4; ++i)
      b0[i] = load_b_frag(brow0 + (size_t)i * 16 * ldb + kk, hi);
#pragma unroll
    for (int i = 0; i < 4; ++i) {
      acc[i]     = wmma_bf16(a1lo, b1[i], acc[i]);
      acc[4 + i] = wmma_bf16(a1hi, b1[i], acc[4 + i]);
    }
    a1lo = load_a_frag(arow0 + kk + 32, hi);
    a1hi = load_a_frag(arow1 + kk + 32, hi);
#pragma unroll
    for (int i = 0; i < 4; ++i)
      b1[i] = load_b_frag(brow0 + (size_t)i * 16 * ldb + kk + 32, hi);
  }
#pragma unroll
  for (int i = 0; i < 4; ++i) {
    acc[i]     = wmma_bf16(a0lo, b0[i], acc[i]);
    acc[4 + i] = wmma_bf16(a0hi, b0[i], acc[4 + i]);
  }
#pragma unroll
  for (int i = 0; i < 4; ++i) {
    acc[i]     = wmma_bf16(a1lo, b1[i], acc[i]);
    acc[4 + i] = wmma_bf16(a1hi, b1[i], acc[4 + i]);
  }
}

// ---------------- 1) fp32 -> bf16 conversion ----------------
struct alignas(8) bf16x4 { bf16_t v[4]; };

__global__ void cvt_kernel(const float* __restrict__ in, bf16_t* __restrict__ out,
                           int n4) {
  int i = blockIdx.x * 256 + threadIdx.x;
  if (i >= n4) return;
  float4 f = ((const float4*)in)[i];
  bf16x4 o;
  o.v[0] = to_bf16(f.x); o.v[1] = to_bf16(f.y);
  o.v[2] = to_bf16(f.z); o.v[3] = to_bf16(f.w);
  ((bf16x4*)out)[i] = o;
}

// ---------------- 2) QKV projection ----------------
// grid: (S/64, D/256, B*H), block 256 (8 waves: 2(M)x4(N), wave tile 32x64)
// q,k: [B,H,S,D] bf16 (+bias);  vt: [B,H,D,S] bf16 (+bias), transposed for PV.
__global__ __launch_bounds__(256) void qkv_kernel(
    const bf16_t* __restrict__ x,                        // [B,S,D] bf16
    const bf16_t* __restrict__ wq, const bf16_t* __restrict__ wk,
    const bf16_t* __restrict__ wv,                       // [H,D,D] bf16
    const float* __restrict__ bq, const float* __restrict__ bk,
    const float* __restrict__ bv,                        // [H,D] f32
    bf16_t* __restrict__ q, bf16_t* __restrict__ k, bf16_t* __restrict__ vt) {
  const int bh = blockIdx.z, b = bh / CH, h = bh % CH;
  const int lane = threadIdx.x & 31, wave = threadIdx.x >> 5;
  const int hi = lane >> 4, ln = lane & 15;
  const int m0 = blockIdx.x * 64 + (wave & 1) * 32;      // S rows (32 per wave)
  const int n0 = blockIdx.y * 256 + (wave >> 1) * 64;    // out cols

  const bf16_t* arow0 = x + ((size_t)b * CS + m0 + ln) * CD;
  const bf16_t* arow1 = arow0 + (size_t)16 * CD;
  const size_t wofs = (size_t)h * CD * CD + (size_t)(n0 + ln) * CD;

  // ---- Q ----
  {
    v8f acc[8] = {};
    gemm32x64(arow0, arow1, wq + wofs, CD, CD, hi, acc);
    bf16_t* out = q + (size_t)bh * CS * CD;
#pragma unroll
    for (int i = 0; i < 4; ++i) {
      int n = n0 + i * 16 + ln;
      float bias = bq[h * CD + n];
#pragma unroll
      for (int j = 0; j < 8; ++j) {
        out[(size_t)(m0 + j + 8 * hi) * CD + n]      = to_bf16(acc[i][j] + bias);
        out[(size_t)(m0 + 16 + j + 8 * hi) * CD + n] = to_bf16(acc[4 + i][j] + bias);
      }
    }
  }
  // ---- K ----
  {
    v8f acc[8] = {};
    gemm32x64(arow0, arow1, wk + wofs, CD, CD, hi, acc);
    bf16_t* out = k + (size_t)bh * CS * CD;
#pragma unroll
    for (int i = 0; i < 4; ++i) {
      int n = n0 + i * 16 + ln;
      float bias = bk[h * CD + n];
#pragma unroll
      for (int j = 0; j < 8; ++j) {
        out[(size_t)(m0 + j + 8 * hi) * CD + n]      = to_bf16(acc[i][j] + bias);
        out[(size_t)(m0 + 16 + j + 8 * hi) * CD + n] = to_bf16(acc[4 + i][j] + bias);
      }
    }
  }
  // ---- V (stored transposed: vt[bh][d][s]) ----
  {
    v8f acc[8] = {};
    gemm32x64(arow0, arow1, wv + wofs, CD, CD, hi, acc);
    bf16_t* out = vt + (size_t)bh * CD * CS;
#pragma unroll
    for (int i = 0; i < 4; ++i) {
      int n = n0 + i * 16 + ln;
      float bias = bv[h * CD + n];
#pragma unroll
      for (int j = 0; j < 8; ++j) {
        out[(size_t)n * CS + (m0 + j + 8 * hi)]      = to_bf16(acc[i][j] + bias);
        out[(size_t)n * CS + (m0 + 16 + j + 8 * hi)] = to_bf16(acc[4 + i][j] + bias);
      }
    }
  }
}

// ---------------- 3) attention: scores -> causal softmax -> PV ----------------
// grid: (S/128, B*H), block 256 (8 waves). Wave w owns query rows r0..r0+15 and
// ALL 256 key columns (16 accumulators) -> softmax entirely in-wave.
__global__ __launch_bounds__(256) void attn_kernel(
    const bf16_t* __restrict__ q, const bf16_t* __restrict__ k,
    const bf16_t* __restrict__ vt, bf16_t* __restrict__ cat) {
  __shared__ __align__(16) bf16_t lds_p[8 * 16 * 256];   // 64 KB, P in A-layout

  const int bh = blockIdx.y, bidx = bh / CH, h = bh % CH;
  const int lane = threadIdx.x & 31, wave = threadIdx.x >> 5;
  const int hi = lane >> 4, ln = lane & 15;
  const int r0 = blockIdx.x * 128 + wave * 16;           // query row base

  const bf16_t* Q  = q  + (size_t)bh * CS * CD;
  const bf16_t* Kp = k  + (size_t)bh * CS * CD;
  const bf16_t* V  = vt + (size_t)bh * CD * CS;          // [D][S]

  // ---- phase A: scores[16 x 256] = Q_tile @ K^T ----
  // k-chunks processed in pairs (even: aE, odd: aO); 4 accumulator groups per
  // chunk consume bb0/bb1 alternately; each buffer is refilled right after its
  // wmma with the group used two steps later -> no register moves.
  v8f sc[16] = {};
  const bf16_t* qrow = Q + (size_t)(r0 + ln) * CD;
  v16bf aE = load_a_frag(qrow, hi);
  v16bf aO = load_a_frag(qrow + 32, hi);
  v16bf bb0[4], bb1[4];
#pragma unroll
  for (int i = 0; i < 4; ++i)
    bb0[i] = load_b_frag(Kp + (size_t)(i * 16 + ln) * CD, hi);        // (c0,g0)
#pragma unroll
  for (int i = 0; i < 4; ++i)
    bb1[i] = load_b_frag(Kp + (size_t)((4 + i) * 16 + ln) * CD, hi);  // (c0,g1)
#pragma unroll 1
  for (int kk = 0; kk < CD; kk += 64) {
    const bool more = (kk + 64) < CD;
    // ---- even chunk (kk), A = aE ----
#pragma unroll
    for (int i = 0; i < 4; ++i) sc[i] = wmma_bf16(aE, bb0[i], sc[i]);
#pragma unroll
    for (int i = 0; i < 4; ++i)                                        // (c0,g2)
      bb0[i] = load_b_frag(Kp + (size_t)((8 + i) * 16 + ln) * CD + kk, hi);
#pragma unroll
    for (int i = 0; i < 4; ++i) sc[4 + i] = wmma_bf16(aE, bb1[i], sc[4 + i]);
#pragma unroll
    for (int i = 0; i < 4; ++i)                                        // (c0,g3)
      bb1[i] = load_b_frag(Kp + (size_t)((12 + i) * 16 + ln) * CD + kk, hi);
#pragma unroll
    for (int i = 0; i < 4; ++i) sc[8 + i] = wmma_bf16(aE, bb0[i], sc[8 + i]);
#pragma unroll
    for (int i = 0; i < 4; ++i)                                        // (c1,g0)
      bb0[i] = load_b_frag(Kp + (size_t)(i * 16 + ln) * CD + kk + 32, hi);
#pragma unroll
    for (int i = 0; i < 4; ++i) sc[12 + i] = wmma_bf16(aE, bb1[i], sc[12 + i]);
#pragma unroll
    for (int i = 0; i < 4; ++i)                                        // (c1,g1)
      bb1[i] = load_b_frag(Kp + (size_t)((4 + i) * 16 + ln) * CD + kk + 32, hi);
    // ---- odd chunk (kk+32), A = aO; refill aE/aO for next pair ----
#pragma unroll
    for (int i = 0; i < 4; ++i) sc[i] = wmma_bf16(aO, bb0[i], sc[i]);
#pragma unroll
    for (int i = 0; i < 4; ++i)                                        // (c1,g2)
      bb0[i] = load_b_frag(Kp + (size_t)((8 + i) * 16 + ln) * CD + kk + 32, hi);
    if (more) aE = load_a_frag(qrow + kk + 64, hi);
#pragma unroll
    for (int i = 0; i < 4; ++i) sc[4 + i] = wmma_bf16(aO, bb1[i], sc[4 + i]);
#pragma unroll
    for (int i = 0; i < 4; ++i)                                        // (c1,g3)
      bb1[i] = load_b_frag(Kp + (size_t)((12 + i) * 16 + ln) * CD + kk + 32, hi);
#pragma unroll
    for (int i = 0; i < 4; ++i) sc[8 + i] = wmma_bf16(aO, bb0[i], sc[8 + i]);
    if (more) {
#pragma unroll
      for (int i = 0; i < 4; ++i)                                      // (c0',g0)
        bb0[i] = load_b_frag(Kp + (size_t)(i * 16 + ln) * CD + kk + 64, hi);
    }
#pragma unroll
    for (int i = 0; i < 4; ++i) sc[12 + i] = wmma_bf16(aO, bb1[i], sc[12 + i]);
    if (more) {
#pragma unroll
      for (int i = 0; i < 4; ++i)                                      // (c0',g1)
        bb1[i] = load_b_frag(Kp + (size_t)((4 + i) * 16 + ln) * CD + kk + 64, hi);
      aO = load_a_frag(qrow + kk + 96, hi);
    }
  }

  // ---- causal mask + scale + softmax (per row, in-wave) ----
  const float scale = 0.04419417382415922f;              // 1/sqrt(512)
  bf16_t* Prow = lds_p + wave * 16 * 256;
#pragma unroll
  for (int j = 0; j < 8; ++j) {
    const int srow = r0 + j + 8 * hi;                    // this half's query row
    float vals[16];
    float m = -__builtin_inff();
#pragma unroll
    for (int n = 0; n < 16; ++n) {
      int t = n * 16 + ln;
      float v = sc[n][j] * scale;
      v = (t <= srow) ? v : -__builtin_inff();
      vals[n] = v;
      m = fmaxf(m, v);
    }
#pragma unroll
    for (int off = 1; off < 16; off <<= 1)               // reduce within 16-lane half
      m = fmaxf(m, __shfl_xor(m, off, 32));
    float ssum = 0.f;
#pragma unroll
    for (int n = 0; n < 16; ++n) {
      float e = __expf(vals[n] - m);
      vals[n] = e;
      ssum += e;
    }
#pragma unroll
    for (int off = 1; off < 16; off <<= 1)
      ssum += __shfl_xor(ssum, off, 32);
    float inv = 1.0f / ssum;
#pragma unroll
    for (int n = 0; n < 16; ++n)
      Prow[(j + 8 * hi) * 256 + n * 16 + ln] = to_bf16(vals[n] * inv);
  }
  __syncthreads();

  // ---- phase B: out[16 x 512] = P @ V  (B-fragments from vt rows) ----
#pragma unroll 1
  for (int nc = 0; nc < 8; ++nc) {
    const int d0 = nc * 64;
    v8f acc[4] = {};
    gemm16x64(Prow + (size_t)ln * 256, V + (size_t)(d0 + ln) * CS, CS, CS, hi, acc);
#pragma unroll
    for (int i = 0; i < 4; ++i) {
      int d = d0 + i * 16 + ln;
#pragma unroll
      for (int j = 0; j < 8; ++j) {
        int srow = r0 + j + 8 * hi;
        cat[((size_t)bidx * CS + srow) * CHD + h * CD + d] = to_bf16(acc[i][j]);
      }
    }
  }
}

// ---------------- 4) output projection: out = cat @ Wp^T + bp ----------------
// grid: (B*S/64, D/256), block 256 (8 waves: 2(M)x4(N), wave tile 32x64)
__global__ __launch_bounds__(256) void proj_kernel(
    const bf16_t* __restrict__ cat, const bf16_t* __restrict__ wp,
    const float* __restrict__ bp, float* __restrict__ out) {
  const int lane = threadIdx.x & 31, wave = threadIdx.x >> 5;
  const int hi = lane >> 4, ln = lane & 15;
  const int m0 = blockIdx.x * 64 + (wave & 1) * 32;      // over B*S
  const int n0 = blockIdx.y * 256 + (wave >> 1) * 64;    // over D

  const bf16_t* arow0 = cat + (size_t)(m0 + ln) * CHD;
  const bf16_t* arow1 = arow0 + (size_t)16 * CHD;
  v8f acc[8] = {};
  gemm32x64(arow0, arow1, wp + (size_t)(n0 + ln) * CHD, CHD, CHD, hi, acc);
#pragma unroll
  for (int i = 0; i < 4; ++i) {
    int n = n0 + i * 16 + ln;
    float bias = bp[n];
#pragma unroll
    for (int j = 0; j < 8; ++j) {
      out[(size_t)(m0 + j + 8 * hi) * CD + n]      = acc[i][j] + bias;
      out[(size_t)(m0 + 16 + j + 8 * hi) * CD + n] = acc[4 + i][j] + bias;
    }
  }
}

// ---------------- host launcher ----------------
extern "C" void kernel_launch(void* const* d_in, const int* in_sizes, int n_in,
                              void* d_out, int out_size, void* d_ws, size_t ws_size,
                              hipStream_t stream) {
  const float* x  = (const float*)d_in[0];
  const float* Wq = (const float*)d_in[1];
  const float* bq = (const float*)d_in[2];
  const float* Wk = (const float*)d_in[3];
  const float* bk = (const float*)d_in[4];
  const float* Wv = (const float*)d_in[5];
  const float* bv = (const float*)d_in[6];
  const float* Wp = (const float*)d_in[7];
  const float* bp = (const float*)d_in[8];
  float* out = (float*)d_out;

  char* ws = (char*)d_ws;
  size_t off = 0;
  auto take = [&](size_t elems) -> bf16_t* {
    bf16_t* p = (bf16_t*)(ws + off);
    off += ((elems * sizeof(bf16_t)) + 255) & ~(size_t)255;
    return p;
  };
  bf16_t* x_bf  = take((size_t)CB * CS * CD);
  bf16_t* wq_bf = take((size_t)CH * CD * CD);
  bf16_t* wk_bf = take((size_t)CH * CD * CD);
  bf16_t* wv_bf = take((size_t)CH * CD * CD);
  bf16_t* wp_bf = take((size_t)CD * CHD);
  bf16_t* q_bf  = take((size_t)CB * CH * CS * CD);
  bf16_t* k_bf  = take((size_t)CB * CH * CS * CD);
  bf16_t* vt_bf = take((size_t)CB * CH * CD * CS);
  bf16_t* cat_bf = take((size_t)CB * CS * CHD);

  auto cvt = [&](const float* src, bf16_t* dst, size_t n) {
    int n4 = (int)(n / 4);
    cvt_kernel<<<(n4 + 255) / 256, 256, 0, stream>>>(src, dst, n4);
  };
  cvt(x,  x_bf,  (size_t)CB * CS * CD);
  cvt(Wq, wq_bf, (size_t)CH * CD * CD);
  cvt(Wk, wk_bf, (size_t)CH * CD * CD);
  cvt(Wv, wv_bf, (size_t)CH * CD * CD);
  cvt(Wp, wp_bf, (size_t)CD * CHD);

  qkv_kernel<<<dim3(CS / 64, CD / 256, CB * CH), 256, 0, stream>>>(
      x_bf, wq_bf, wk_bf, wv_bf, bq, bk, bv, q_bf, k_bf, vt_bf);

  attn_kernel<<<dim3(CS / 128, CB * CH), 256, 0, stream>>>(
      q_bf, k_bf, vt_bf, cat_bf);

  proj_kernel<<<dim3((CB * CS) / 64, CD / 256), 256, 0, stream>>>(
      cat_bf, wp_bf, bp, out);
}
